// GNN_Combo_60868276519665
// MI455X (gfx1250) — matrix operand order repaired and checked by reading, not compiled, not run on hardware
//
#include <hip/hip_runtime.h>

typedef __attribute__((ext_vector_type(16))) _Float16 v16h;
typedef __attribute__((ext_vector_type(8)))  _Float16 v8h;
typedef __attribute__((ext_vector_type(8)))  float    v8f;
typedef __attribute__((ext_vector_type(4))) unsigned int u32x4;
typedef __attribute__((ext_vector_type(8))) int i32x8;
typedef __attribute__((ext_vector_type(4))) int i32x4;

// ---------------------------------------------------------------------------
// TDM availability (probe-verified: builtin exists on both toolchains;
// ROCm 7.2 clang-22 takes 5 args, amdgpu-toolchain clang-23 takes 6).
// ---------------------------------------------------------------------------
#if defined(__has_builtin)
#if __has_builtin(__builtin_amdgcn_tensor_load_to_lds) && __has_builtin(__builtin_amdgcn_s_wait_tensorcnt)
#define HAVE_TDM 1
#endif
#endif
#ifndef HAVE_TDM
#define HAVE_TDM 0
#endif

#if HAVE_TDM
// Issue a TDM load of a 2-D f16 tile (rows x cols, row stride in elements)
// from global memory into LDS.  Descriptor packed per CDNA5 D# spec
// (group0: count/lds_addr/global_addr/type=2; group1: data_size=2B,
// tensor dims, tile dims, dim0 stride).  Wave-level op; EXEC ignored.
__device__ __forceinline__ void tdm_load_tile_f16(const _Float16* gsrc, void* lds_dst,
                                                  unsigned rows, unsigned cols,
                                                  unsigned row_stride_elems) {
  unsigned long long ga = (unsigned long long)gsrc;
  unsigned lds_off = (unsigned)(unsigned long long)lds_dst;  // generic addr low 32b == LDS offset
  u32x4 g0;
  g0[0] = 1u;                                 // count=1, user descriptor, no gather
  g0[1] = lds_off;                            // D#.lds_addr
  g0[2] = (unsigned)ga;                       // global_addr[31:0]
  g0[3] = (unsigned)(ga >> 32) | (2u << 30);  // global_addr[56:32] | type=2 (image)
  i32x8 g1;
  g1[0] = 1 << 16;                                 // workgroup_mask=0, data_size=1 (2 bytes)
  g1[1] = (int)(cols << 16);                       // tensor_dim0[15:0]   (bits 79:48)
  g1[2] = (int)((cols >> 16) | (rows << 16));      // dim0 hi | tensor_dim1 lo
  g1[3] = (int)((rows >> 16) | (cols << 16));      // dim1 hi | tile_dim0
  g1[4] = (int)rows;                               // tile_dim1 (tile_dim2 = 0)
  g1[5] = (int)row_stride_elems;                   // tensor_dim0_stride lo
  g1[6] = 0;
  g1[7] = 0;
  i32x4 gz = {0, 0, 0, 0};
#if __clang_major__ >= 23
  i32x8 gz8 = {0, 0, 0, 0, 0, 0, 0, 0};
  __builtin_amdgcn_tensor_load_to_lds(g0, g1, gz, gz, gz8, 0);
#else
  __builtin_amdgcn_tensor_load_to_lds(g0, g1, gz, gz, 0);
#endif
}
#endif  // HAVE_TDM

// ---------------------------------------------------------------------------
// WMMA helpers (CDNA5 wave32, 16x16x32 f16 -> f32)
// ---------------------------------------------------------------------------
__device__ __forceinline__ v8f wmma_f16(v16h a, v16h b, v8f c) {
  return __builtin_amdgcn_wmma_f32_16x16x32_f16(
      /*neg_a=*/false, a, /*neg_b=*/false, b,
      /*c_mod=*/(short)0, c, /*reuse_a=*/false, /*reuse_b=*/false);
}

// A-matrix 16x32 f16 fragment (contiguous k = 16*half + s, matching B): p0 points at (row0, k0).
__device__ __forceinline__ v16h load_A_frag(const _Float16* p0, int ld, int lane) {
  const int m  = lane & 15;
  const int kb = (lane & 16) ? 16 : 0;
  const _Float16* r = p0 + m * ld;
  v16h a;
#pragma unroll
  for (int v = 0; v < 8; ++v) {
    const int kk = kb + 2 * v;
    a[2 * v]     = r[kk];
    a[2 * v + 1] = r[kk + 1];
  }
  return a;
}

// B-matrix 32x16 f16 fragment from K-major (row = k) source at (k0, col0).
__device__ __forceinline__ v16h load_B_frag(const _Float16* p0, int ld, int lane) {
  const int n  = lane & 15;
  const int kb = (lane & 16) ? 16 : 0;
  v16h b;
#pragma unroll
  for (int v = 0; v < 8; ++v) {
    const int kk = kb + 2 * v;
    b[2 * v]     = p0[kk * ld + n];
    b[2 * v + 1] = p0[(kk + 1) * ld + n];
  }
  return b;
}

// B-matrix 32x16 fragment from N-major source: element (k,n) = p0[n*ld + k].
__device__ __forceinline__ v16h load_B_fragT(const _Float16* p0, int ld, int lane) {
  const int n  = lane & 15;
  const int kb = (lane & 16) ? 16 : 0;
  const _Float16* r = p0 + n * ld;
  v16h b;
#pragma unroll
  for (int v = 0; v < 8; ++v) {
    const int kk = kb + 2 * v;
    b[2 * v]     = r[kk];
    b[2 * v + 1] = r[kk + 1];
  }
  return b;
}

// ---------------------------------------------------------------------------
// Prep: x (f32) -> xh (f16) + squared norms (f32)
// ---------------------------------------------------------------------------
__global__ void prep_x_kernel(const float* __restrict__ x,
                              _Float16* __restrict__ xh,
                              float* __restrict__ sq, int N) {
  int i = blockIdx.x * blockDim.x + threadIdx.x;
  if (i >= N) return;
  float acc = 0.0f;
#pragma unroll
  for (int c = 0; c < 64; ++c) {
    float v = x[i * 64 + c];
    xh[i * 64 + c] = (_Float16)v;
    acc += v * v;
  }
  sq[i] = acc;
}

__global__ void cvt_f16_kernel(const float* __restrict__ src,
                               _Float16* __restrict__ dst, int n) {
  int t = blockIdx.x * blockDim.x + threadIdx.x;
  if (t < n) dst[t] = (_Float16)src[t];
}

// ---------------------------------------------------------------------------
// kNN: WMMA Gram tiles + streaming register-resident top-16 per query.
// Block = 128 threads (4 waves); wave w owns 16 query rows.
// Candidate tiles are double-buffered via TDM (tensor_load_to_lds) so the
// DMA of tile j0+16 overlaps WMMA + selection on tile j0.
// ---------------------------------------------------------------------------
__global__ __launch_bounds__(128) void knn_kernel(const _Float16* __restrict__ xh,
                                                  const float* __restrict__ sq,
                                                  int* __restrict__ knn_idx, int N) {
  __shared__ _Float16 cand[2][16][64];  // double-buffered candidate tile
  __shared__ float    sqc[16];
  __shared__ float    dt[4][16][17];    // per-wave distance tile (padded)

  const int lane = threadIdx.x & 31;
  const int wave = threadIdx.x >> 5;
  const int q0   = blockIdx.x * 64 + wave * 16;
  const int hi8  = (lane & 16) ? 8 : 0;
  const int n16  = lane & 15;

  // Query A-fragments, loaded once (K = 0..31 and 32..63)
  v16h a0 = load_A_frag(xh + q0 * 64 + 0,  64, lane);
  v16h a1 = load_A_frag(xh + q0 * 64 + 32, 64, lane);

  v8f sqq;
#pragma unroll
  for (int v = 0; v < 8; ++v) sqq[v] = sq[q0 + v + hi8];

  float dbest[16];
  int   ibest[16];
#pragma unroll
  for (int s = 0; s < 16; ++s) { dbest[s] = 3.0e38f; ibest[s] = 0; }

#if HAVE_TDM
  if (wave == 0) tdm_load_tile_f16(xh, &cand[0][0][0], 16, 64, 64);  // prologue tile
#endif

  for (int j0 = 0; j0 < N; j0 += 16) {
#if HAVE_TDM
    const int buf = (j0 >> 4) & 1;
#else
    const int buf = 0;
#endif
    __syncthreads();  // previous tile fully consumed by all waves
#if HAVE_TDM
    if (wave == 0) {
      if (j0 + 16 < N) {  // kick next DMA, then wait only for the current tile
        tdm_load_tile_f16(xh + (j0 + 16) * 64, &cand[buf ^ 1][0][0], 16, 64, 64);
        __builtin_amdgcn_s_wait_tensorcnt(1);
      } else {
        __builtin_amdgcn_s_wait_tensorcnt(0);
      }
    }
    if (threadIdx.x < 16) sqc[threadIdx.x] = sq[j0 + threadIdx.x];
#else
    {  // cooperative stage of 16x64 candidate tile (b128 per thread)
      const int r  = threadIdx.x >> 3;
      const int c8 = (threadIdx.x & 7) * 8;
      *(v8h*)&cand[0][r][c8] = *(const v8h*)&xh[(j0 + r) * 64 + c8];
      if (threadIdx.x < 16) sqc[threadIdx.x] = sq[j0 + threadIdx.x];
    }
    if (j0 + 16 < N) __builtin_prefetch(&xh[(j0 + 16) * 64], 0, 1);
#endif
    __syncthreads();  // tile + norms visible to all waves

    v16h b0 = load_B_fragT(&cand[buf][0][0],  64, lane);
    v16h b1 = load_B_fragT(&cand[buf][0][32], 64, lane);
    v8f  c  = {};
    c = wmma_f16(a0, b0, c);
    c = wmma_f16(a1, b1, c);

    const float sqcn = sqc[n16];
#pragma unroll
    for (int v = 0; v < 8; ++v)
      dt[wave][v + hi8][n16] = sqq[v] + sqcn - 2.0f * c[v];
    asm volatile("s_wait_dscnt 0" ::: "memory");

    if (lane < 16) {  // lane == query row within this wave's tile
#pragma unroll
      for (int nn = 0; nn < 16; ++nn) {
        float d = dt[wave][n16][nn];
        if (d < dbest[15]) {
          dbest[15] = d;
          ibest[15] = j0 + nn;
#pragma unroll
          for (int s = 15; s > 0; --s) {  // single bubble-up keeps it sorted
            if (dbest[s] < dbest[s - 1]) {
              float td = dbest[s]; dbest[s] = dbest[s - 1]; dbest[s - 1] = td;
              int   ti = ibest[s]; ibest[s] = ibest[s - 1]; ibest[s - 1] = ti;
            }
          }
        }
      }
    }
  }

  if (lane < 16) {
#pragma unroll
    for (int s = 0; s < 16; ++s)
      knn_idx[(q0 + n16) * 16 + s] = ibest[s];
  }
}

// ---------------------------------------------------------------------------
// EdgeConv: per node, m = [x_i, x_j - x_i] (16x128), two WMMA layers,
// log-softmax per row, max over k.  4 waves/block; W1+W2 staged by TDM.
// ---------------------------------------------------------------------------
#define EC_WAVES 4
#define EC_LDS_BYTES (2 * 16384 * 2 /*W1,W2 f16*/ + EC_WAVES * (2176 + 2176) * 2 /*mA,L1 f16*/ + EC_WAVES * 16 * 132 * 4 /*L2 f32*/)

__global__ __launch_bounds__(128) void edgeconv_kernel(
    const _Float16* __restrict__ xh, const int* __restrict__ knn_idx,
    const _Float16* __restrict__ w1h, const float* __restrict__ b1,
    const _Float16* __restrict__ w2h, const float* __restrict__ b2,
    _Float16* __restrict__ h_out, int N) {
  extern __shared__ unsigned char smem[];
  __shared__ float rowm[EC_WAVES][16];
  __shared__ float rowl[EC_WAVES][16];

  const int lane = threadIdx.x & 31;
  const int wave = threadIdx.x >> 5;
  const int hi8  = (lane & 16) ? 8 : 0;
  const int n16  = lane & 15;

  _Float16* W1 = (_Float16*)smem;          // 128x128
  _Float16* W2 = W1 + 16384;               // 128x128
  _Float16* mA = W2 + 16384 + wave * 2176;                    // [16][136]
  _Float16* L1 = W2 + 16384 + EC_WAVES * 2176 + wave * 2176;  // [16][136]
  float*    L2 = (float*)(smem + (2 * 16384 + 2 * EC_WAVES * 2176) * 2)
                 + wave * (16 * 132);                         // [16][132]

#if HAVE_TDM
  if (wave == 0) {
    tdm_load_tile_f16(w1h, W1, 128, 128, 128);
    tdm_load_tile_f16(w2h, W2, 128, 128, 128);
    __builtin_amdgcn_s_wait_tensorcnt(0);
  }
#else
  for (int t = threadIdx.x; t < 16384 / 8; t += 128) {
    ((v8h*)W1)[t] = ((const v8h*)w1h)[t];
    ((v8h*)W2)[t] = ((const v8h*)w2h)[t];
  }
#endif
  __syncthreads();

  const int i = blockIdx.x * EC_WAVES + wave;

  // build message matrix mA[r][c]: c<64 -> x_i ; c>=64 -> x_j - x_i
  for (int t = lane; t < 16 * 128; t += 32) {
    const int r = t >> 7, c = t & 127;
    const int nb = knn_idx[i * 16 + r];
    _Float16 xi = xh[i * 64 + (c & 63)];
    _Float16 val = (c < 64) ? xi : (_Float16)(xh[nb * 64 + (c - 64)] - xi);
    mA[r * 136 + c] = val;
  }
  asm volatile("s_wait_dscnt 0" ::: "memory");

  // layer 1: relu(m @ W1 + b1) -> L1 (f16)
#pragma unroll 1
  for (int n0 = 0; n0 < 8; ++n0) {
    v8f c = {};
#pragma unroll
    for (int kc = 0; kc < 4; ++kc) {
      v16h a = load_A_frag(mA + kc * 32, 136, lane);
      v16h b = load_B_frag(W1 + (kc * 32) * 128 + n0 * 16, 128, lane);
      c = wmma_f16(a, b, c);
    }
    const float bias = b1[n0 * 16 + n16];
#pragma unroll
    for (int v = 0; v < 8; ++v)
      L1[(v + hi8) * 136 + n0 * 16 + n16] = (_Float16)fmaxf(c[v] + bias, 0.0f);
  }
  asm volatile("s_wait_dscnt 0" ::: "memory");

  // layer 2: L1 @ W2 + b2 -> L2 (f32)
#pragma unroll 1
  for (int n0 = 0; n0 < 8; ++n0) {
    v8f c = {};
#pragma unroll
    for (int kc = 0; kc < 4; ++kc) {
      v16h a = load_A_frag(L1 + kc * 32, 136, lane);
      v16h b = load_B_frag(W2 + (kc * 32) * 128 + n0 * 16, 128, lane);
      c = wmma_f16(a, b, c);
    }
    const float bias = b2[n0 * 16 + n16];
#pragma unroll
    for (int v = 0; v < 8; ++v)
      L2[(v + hi8) * 132 + n0 * 16 + n16] = c[v] + bias;
  }
  asm volatile("s_wait_dscnt 0" ::: "memory");

  // per-row log-softmax stats
  if (lane < 16) {
    const int r = n16;
    float mx = -3.0e38f;
#pragma unroll 1
    for (int c = 0; c < 128; ++c) mx = fmaxf(mx, L2[r * 132 + c]);
    float s = 0.0f;
#pragma unroll 1
    for (int c = 0; c < 128; ++c) s += __expf(L2[r * 132 + c] - mx);
    rowm[wave][r] = mx;
    rowl[wave][r] = __logf(s);
  }
  asm volatile("s_wait_dscnt 0" ::: "memory");

  // max over k per column -> h[i]
#pragma unroll
  for (int cc = 0; cc < 4; ++cc) {
    const int c = lane * 4 + cc;
    float hm = -3.0e38f;
#pragma unroll 1
    for (int r = 0; r < 16; ++r)
      hm = fmaxf(hm, L2[r * 132 + c] - rowm[wave][r] - rowl[wave][r]);
    h_out[i * 128 + c] = (_Float16)hm;
  }
}

// ---------------------------------------------------------------------------
// Generic WMMA GEMM: C[Mx128] = A[Mx128] @ B[128x128]   (A,B f16; C f32)
// B staged into LDS by TDM.
// ---------------------------------------------------------------------------
__global__ __launch_bounds__(256) void gemm128_kernel(const _Float16* __restrict__ A,
                                                      const _Float16* __restrict__ B,
                                                      float* __restrict__ C, int M) {
  __shared__ _Float16 Bs[128 * 128];
  const int lane = threadIdx.x & 31;
  const int wave = threadIdx.x >> 5;
#if HAVE_TDM
  if (wave == 0) {
    tdm_load_tile_f16(B, Bs, 128, 128, 128);
    __builtin_amdgcn_s_wait_tensorcnt(0);
  }
#else
  for (int t = threadIdx.x; t < 128 * 128 / 8; t += 256)
    ((v8h*)Bs)[t] = ((const v8h*)B)[t];
#endif
  __syncthreads();

  const int m0  = (blockIdx.x * 8 + wave) * 16;
  const int hi8 = (lane & 16) ? 8 : 0;
  const int n16 = lane & 15;
  if (m0 >= M) return;

  v16h a[4];
#pragma unroll
  for (int kc = 0; kc < 4; ++kc)
    a[kc] = load_A_frag(A + m0 * 128 + kc * 32, 128, lane);

#pragma unroll 1
  for (int n0 = 0; n0 < 8; ++n0) {
    v8f c = {};
#pragma unroll
    for (int kc = 0; kc < 4; ++kc) {
      v16h b = load_B_frag(Bs + (kc * 32) * 128 + n0 * 16, 128, lane);
      c = wmma_f16(a[kc], b, c);
    }
#pragma unroll
    for (int v = 0; v < 8; ++v)
      C[(m0 + v + hi8) * 128 + n0 * 16 + n16] = c[v];
  }
}

// ---------------------------------------------------------------------------
// GCN aggregate: deg == 17 for every node -> norm == 1/17 everywhere.
// h2[i][c] = (hw[i][c] + sum_j hw[idx[i][j]][c]) / 17 + gcn_b[c]
// ---------------------------------------------------------------------------
__global__ void gcn_gather_kernel(const float* __restrict__ hw,
                                  const int* __restrict__ knn_idx,
                                  const float* __restrict__ gcn_b,
                                  _Float16* __restrict__ h2h, int N) {
  int t = blockIdx.x * blockDim.x + threadIdx.x;
  if (t >= N * 128) return;
  const int i = t >> 7, c = t & 127;
  float acc = hw[i * 128 + c];
#pragma unroll
  for (int s = 0; s < 16; ++s)
    acc += hw[knn_idx[i * 16 + s] * 128 + c];
  h2h[t] = (_Float16)(acc * (1.0f / 17.0f) + gcn_b[c]);
}

// ---------------------------------------------------------------------------
// Row softmax over 128 (+ bias), one wave per row.
// ---------------------------------------------------------------------------
__global__ __launch_bounds__(256) void softmax128_kernel(float* __restrict__ X,
                                                         const float* __restrict__ bias,
                                                         int N) {
  const int lane = threadIdx.x & 31;
  const int wave = threadIdx.x >> 5;
  const int r    = blockIdx.x * 8 + wave;
  if (r >= N) return;
  float v[4];
  float mx = -3.0e38f;
#pragma unroll
  for (int k = 0; k < 4; ++k) {
    const int c = lane + k * 32;
    v[k] = X[r * 128 + c] + bias[c];
    mx = fmaxf(mx, v[k]);
  }
#pragma unroll
  for (int off = 16; off > 0; off >>= 1) mx = fmaxf(mx, __shfl_xor(mx, off, 32));
  float s = 0.0f;
#pragma unroll
  for (int k = 0; k < 4; ++k) { v[k] = __expf(v[k] - mx); s += v[k]; }
#pragma unroll
  for (int off = 16; off > 0; off >>= 1) s += __shfl_xor(s, off, 32);
  const float inv = 1.0f / s;
#pragma unroll
  for (int k = 0; k < 4; ++k) X[r * 128 + lane + k * 32] = v[k] * inv;
}

// ---------------------------------------------------------------------------
// Final head: out = softmax(g @ out_w + out_b), O = 10. One thread per node.
// ---------------------------------------------------------------------------
__global__ void head_kernel(const float* __restrict__ G,
                            const float* __restrict__ W,
                            const float* __restrict__ b,
                            float* __restrict__ out, int N) {
  int i = blockIdx.x * blockDim.x + threadIdx.x;
  if (i >= N) return;
  float acc[10];
#pragma unroll
  for (int o = 0; o < 10; ++o) acc[o] = b[o];
#pragma unroll 1
  for (int k = 0; k < 128; ++k) {
    const float g = G[i * 128 + k];
#pragma unroll
    for (int o = 0; o < 10; ++o) acc[o] += g * W[k * 10 + o];
  }
  float mx = acc[0];
#pragma unroll
  for (int o = 1; o < 10; ++o) mx = fmaxf(mx, acc[o]);
  float s = 0.0f;
#pragma unroll
  for (int o = 0; o < 10; ++o) { acc[o] = __expf(acc[o] - mx); s += acc[o]; }
  const float inv = 1.0f / s;
#pragma unroll
  for (int o = 0; o < 10; ++o) out[i * 10 + o] = acc[o] * inv;
}

// ---------------------------------------------------------------------------
// Host launcher
// ---------------------------------------------------------------------------
extern "C" void kernel_launch(void* const* d_in, const int* in_sizes, int n_in,
                              void* d_out, int out_size, void* d_ws, size_t ws_size,
                              hipStream_t stream) {
  (void)n_in; (void)out_size; (void)ws_size;
  const int N = in_sizes[0] / 64;  // 16384

  const float* x     = (const float*)d_in[0];
  const float* w1    = (const float*)d_in[2];
  const float* b1    = (const float*)d_in[3];
  const float* w2    = (const float*)d_in[4];
  const float* b2    = (const float*)d_in[5];
  const float* gcnw  = (const float*)d_in[6];
  const float* gcnb  = (const float*)d_in[7];
  const float* gow   = (const float*)d_in[8];
  const float* gob   = (const float*)d_in[9];
  const float* outw  = (const float*)d_in[10];
  const float* outb  = (const float*)d_in[11];
  float* out = (float*)d_out;

  // workspace layout (bytes)
  unsigned char* ws = (unsigned char*)d_ws;
  _Float16* xh    = (_Float16*)(ws + 0);                      // N*64 f16
  float*    sq    = (float*)   (ws + 2097152);                // N f32
  int*      idx   = (int*)     (ws + 2162688);                // N*16 i32
  _Float16* w1h   = (_Float16*)(ws + 3211264);                // 128*128 f16
  _Float16* w2h   = (_Float16*)(ws + 3244032);
  _Float16* gcnwh = (_Float16*)(ws + 3276800);
  _Float16* gowh  = (_Float16*)(ws + 3309568);
  _Float16* h     = (_Float16*)(ws + 3342336);                // N*128 f16
  float*    hw    = (float*)   (ws + 7536640);                // N*128 f32
  _Float16* h2h   = (_Float16*)(ws + 15925248);               // N*128 f16
  float*    g     = (float*)   (ws + 20119552);               // N*128 f32

  // prep / conversions
  prep_x_kernel<<<(N + 255) / 256, 256, 0, stream>>>(x, xh, sq, N);
  cvt_f16_kernel<<<64, 256, 0, stream>>>(w1,   w1h,   16384);
  cvt_f16_kernel<<<64, 256, 0, stream>>>(w2,   w2h,   16384);
  cvt_f16_kernel<<<64, 256, 0, stream>>>(gcnw, gcnwh, 16384);
  cvt_f16_kernel<<<64, 256, 0, stream>>>(gow,  gowh,  16384);

  // dynamic kNN graph
  knn_kernel<<<N / 64, 128, 0, stream>>>(xh, sq, idx, N);

  // EdgeConv (needs >64KB dynamic LDS; 320KB/WGP on CDNA5)
  (void)hipFuncSetAttribute((const void*)edgeconv_kernel,
                            hipFuncAttributeMaxDynamicSharedMemorySize,
                            EC_LDS_BYTES);
  edgeconv_kernel<<<N / EC_WAVES, 128, EC_LDS_BYTES, stream>>>(
      xh, idx, w1h, b1, w2h, b2, h, N);

  // GCN: hw = h @ gcn_w ; aggregate (1/17) + bias -> h2h (f16)
  gemm128_kernel<<<N / 128, 256, 0, stream>>>(h, gcnwh, hw, N);
  gcn_gather_kernel<<<(N * 128) / 256, 256, 0, stream>>>(hw, idx, gcnb, h2h, N);

  // inner head: softmax(h2 @ gnn_out_w + b)
  gemm128_kernel<<<N / 128, 256, 0, stream>>>(h2h, gowh, g, N);
  softmax128_kernel<<<N / 8, 256, 0, stream>>>(g, gob, N);

  // final head: softmax(g @ out_w + out_b) -> [N,10]
  head_kernel<<<(N + 255) / 256, 256, 0, stream>>>(g, outw, outb, out, N);
}
